// TransformerEncoderLayer_23141283790917
// MI455X (gfx1250) — matrix operand order repaired and checked by reading, not compiled, not run on hardware
//
#include <hip/hip_runtime.h>
#include <hip/hip_bf16.h>

#define Bn 2
#define Sn 256
#define Hn 768
#define NHh 12
#define Dh 64
#define BS (Bn * Sn)

typedef __attribute__((ext_vector_type(16))) _Float16 v16h;
typedef __attribute__((ext_vector_type(8)))  float    v8f;
typedef __attribute__((ext_vector_type(8)))  _Float16 h8;
typedef __attribute__((ext_vector_type(4)))  float    f4;

__device__ __forceinline__ v8f wmma_f16(v16h a, v16h b, v8f c) {
    return __builtin_amdgcn_wmma_f32_16x16x32_f16(false, a, false, b, (short)0, c, false, false);
}

// A-fragment (16x32 MxK): lane 0-15 -> M=lane, K=0..15 ; lane 16-31 -> M=lane-16, K=16..31
__device__ __forceinline__ v16h load_a(const _Float16* base, int stride, int row, int kbase, int lane) {
    const int r  = row + (lane & 15);
    const int k0 = kbase + ((lane >> 4) << 4);
    const _Float16* p = base + (size_t)r * stride + k0;
    h8 lo = *(const h8*)(p);
    h8 hi = *(const h8*)(p + 8);
    v16h out;
#pragma unroll
    for (int i = 0; i < 8; ++i) { out[i] = lo[i]; out[i + 8] = hi[i]; }
    return out;
}

// A-fragment with per-lane row base already resolved (for batched layouts)
__device__ __forceinline__ v16h load_a_base(const _Float16* rowbase, int kbase, int lane) {
    const int k0 = kbase + ((lane >> 4) << 4);
    const _Float16* p = rowbase + k0;
    h8 lo = *(const h8*)(p);
    h8 hi = *(const h8*)(p + 8);
    v16h out;
#pragma unroll
    for (int i = 0; i < 8; ++i) { out[i] = lo[i]; out[i + 8] = hi[i]; }
    return out;
}

// B-fragment (32x16 KxN), memory stored n-major (row = n, K contiguous):
// lane 0-15 -> N=lane, K=0..15 ; lane 16-31 -> N=lane-16, K=16..31
__device__ __forceinline__ v16h load_b(const _Float16* base, int stride, int col, int kbase, int lane) {
    const int n  = col + (lane & 15);
    const int k0 = kbase + ((lane >> 4) << 4);
    return *(const v16h*)(base + (size_t)n * stride + k0);
}

// fp32 source, converted on the fly, loaded NON-TEMPORAL: pos_embedding (402 MB) is
// streamed exactly once -> keep it from evicting the hot ~30 MB working set in L2.
__device__ __forceinline__ v16h load_b_f32_nt(const float* base, int stride, int col, int kbase, int lane) {
    const int n  = col + (lane & 15);
    const int k0 = kbase + ((lane >> 4) << 4);
    const float* p = base + (size_t)n * stride + k0;
    v16h out;
#pragma unroll
    for (int i = 0; i < 4; ++i) {
        f4 x = __builtin_nontemporal_load((const f4*)(p + 4 * i));
        out[4 * i + 0] = (_Float16)x[0];
        out[4 * i + 1] = (_Float16)x[1];
        out[4 * i + 2] = (_Float16)x[2];
        out[4 * i + 3] = (_Float16)x[3];
    }
    return out;
}

// ---------------- conversion kernels ----------------
__global__ void k_conv(const float* __restrict__ src, _Float16* __restrict__ dst, int n) {
    int i = blockIdx.x * 256 + threadIdx.x;
    if (i < n) dst[i] = (_Float16)src[i];
}

__global__ void k_transconv(const float* __restrict__ src, _Float16* __restrict__ dst) {
    int idx = blockIdx.x * 256 + threadIdx.x;
    if (idx < Hn * Hn) {
        int o = idx / Hn, i = idx - o * Hn;      // dst[o][i] = src[i][o]
        dst[idx] = (_Float16)src[(size_t)i * Hn + o];
    }
}

// ---------------- QKV projection (M=512, N=768, K=768) ----------------
__global__ __launch_bounds__(32) void k_qkv(
    const _Float16* __restrict__ x16,
    const _Float16* __restrict__ wqT, const _Float16* __restrict__ wkT, const _Float16* __restrict__ wvT,
    const float* __restrict__ bq, const float* __restrict__ bk, const float* __restrict__ bv,
    const float* __restrict__ u, const float* __restrict__ v,
    _Float16* __restrict__ qu16, _Float16* __restrict__ qv16,
    _Float16* __restrict__ kk16, _Float16* __restrict__ vvT16) {
    const int lane = threadIdx.x;
    const int mt = blockIdx.x, nt = blockIdx.y, z = blockIdx.z;
    const _Float16* wT   = (z == 0) ? wqT : (z == 1) ? wkT : wvT;
    const float*    bias = (z == 0) ? bq  : (z == 1) ? bk  : bv;
    v8f acc = {};
    for (int kk = 0; kk < Hn; kk += 32) {
        v16h a = load_a(x16, Hn, mt * 16, kk, lane);
        v16h b = load_b(wT,  Hn, nt * 16, kk, lane);
        acc = wmma_f16(a, b, acc);
    }
    const int n = nt * 16 + (lane & 15);
    const int h = n >> 6, dd = n & 63;
    const int mb = (lane >> 4) * 8;
#pragma unroll
    for (int j = 0; j < 8; ++j) {
        const int m = mt * 16 + mb + j;
        const int bb = m >> 8, s = m & 255;
        const float val = acc[j] + bias[n];
        const size_t idx = ((size_t)(bb * NHh + h) * Sn + s) * Dh + dd;
        if (z == 0) {
            qu16[idx] = (_Float16)(val + u[n]);
            qv16[idx] = (_Float16)(val + v[n]);
        } else if (z == 1) {
            kk16[idx] = (_Float16)val;
        } else {
            vvT16[((size_t)(bb * NHh + h) * Dh + dd) * Sn + s] = (_Float16)val;
        }
    }
}

// ---------------- w = (q+v) @ Wr^T-per-head (per head: M=512, N=768, K=64) ----------------
__global__ __launch_bounds__(32) void k_wproj(const _Float16* __restrict__ qv16,
                                              const _Float16* __restrict__ wr16,
                                              _Float16* __restrict__ w16) {
    const int lane = threadIdx.x;
    const int mt = blockIdx.x, nt = blockIdx.y, h = blockIdx.z;
    const int m  = mt * 16 + (lane & 15);
    const int bb = m >> 8, q = m & 255;
    const _Float16* abase = qv16 + ((size_t)(bb * NHh + h) * Sn + q) * Dh;
    v8f acc = {};
#pragma unroll
    for (int kk = 0; kk < Dh; kk += 32) {
        v16h a = load_a_base(abase, kk, lane);
        v16h b = load_b(wr16 + h * Dh, Hn, nt * 16, kk, lane);  // B[dd,c] = Wr[c, h*64+dd]
        acc = wmma_f16(a, b, acc);
    }
    const int c  = nt * 16 + (lane & 15);
    const int mb = (lane >> 4) * 8;
#pragma unroll
    for (int j = 0; j < 8; ++j) {
        const int mm = mt * 16 + mb + j;
        const int b2 = mm >> 8, q2 = mm & 255;
        w16[((size_t)(b2 * NHh + h) * Sn + q2) * Hn + c] = (_Float16)acc[j];
    }
}

// ---------------- A_C = (q+u) k^T per (b,h)  (M=256, N=256, K=64) ----------------
__global__ __launch_bounds__(32) void k_ac(const _Float16* __restrict__ qu16,
                                           const _Float16* __restrict__ kk16,
                                           float* __restrict__ acS) {
    const int lane = threadIdx.x;
    const int qt = blockIdx.x, kt = blockIdx.y, bh = blockIdx.z;
    const _Float16* qb = qu16 + (size_t)bh * Sn * Dh;
    const _Float16* kb = kk16 + (size_t)bh * Sn * Dh;
    v8f acc = {};
#pragma unroll
    for (int kk = 0; kk < Dh; kk += 32) {
        v16h a = load_a(qb, Dh, qt * 16, kk, lane);
        v16h b = load_b(kb, Dh, kt * 16, kk, lane);
        acc = wmma_f16(a, b, acc);
    }
    const int n = kt * 16 + (lane & 15);
    const int mb = (lane >> 4) * 8;
#pragma unroll
    for (int j = 0; j < 8; ++j) {
        const int m = qt * 16 + mb + j;
        acS[(size_t)bh * Sn * Sn + (size_t)m * Sn + n] = acc[j];
    }
}

// ---------------- B_D + scores + softmax, one workgroup per (b,q) ----------------
// Streams pos[b,q,:,:] (256x768 fp32) exactly once with NT loads: the HBM-roofline term.
__global__ __launch_bounds__(256) void k_bd_softmax(const _Float16* __restrict__ w16,
                                                    const float* __restrict__ pos,
                                                    const float* __restrict__ acS,
                                                    const unsigned char* __restrict__ mask,
                                                    _Float16* __restrict__ attn16) {
    __shared__ float sc[16][Sn];
    const int tid = threadIdx.x;
    const int lane = tid & 31, wv = tid >> 5;
    const int q = blockIdx.x, bb = blockIdx.y;
    const float* pbase = pos + ((size_t)bb * Sn + q) * Sn * Hn;
    int hh = lane & 15; if (hh > NHh - 1) hh = NHh - 1;   // pad rows read head 11, output ignored
    const _Float16* abase = w16 + ((size_t)(bb * NHh + hh) * Sn + q) * Hn;

    for (int t = 0; t < 2; ++t) {
        const int kt = wv * 2 + t;           // 8 waves x 2 = 16 key tiles
        v8f acc = {};
        for (int kk = 0; kk < Hn; kk += 32) {
            v16h a = load_a_base(abase, kk, lane);
            v16h b = load_b_f32_nt(pbase, Hn, kt * 16, kk, lane);
            acc = wmma_f16(a, b, acc);
        }
        const int n = kt * 16 + (lane & 15);
        const int mb = (lane >> 4) * 8;
        const bool mk = mask[bb * Sn + n] != 0;
#pragma unroll
        for (int j = 0; j < 8; ++j) {
            const int m = mb + j;            // head index
            if (m < NHh) {
                float s = (acS[((size_t)(bb * NHh + m) * Sn + q) * Sn + n] + acc[j]) * 0.125f;
                sc[m][n] = mk ? s : -1e15f;
            }
        }
    }
    __syncthreads();

    // softmax: wave wv -> row wv, and waves 0..3 -> rows 8..11
    for (int rr = 0; rr < 2; ++rr) {
        const int row = wv + rr * 8;
        if (row >= NHh) continue;
        float e[8];
        float mx = -3.0e38f;
#pragma unroll
        for (int i = 0; i < 8; ++i) { float x = sc[row][lane + 32 * i]; e[i] = x; mx = fmaxf(mx, x); }
#pragma unroll
        for (int o = 16; o >= 1; o >>= 1) mx = fmaxf(mx, __shfl_xor(mx, o, 32));
        float sum = 0.f;
#pragma unroll
        for (int i = 0; i < 8; ++i) { float ex = __expf(e[i] - mx); e[i] = ex; sum += ex; }
#pragma unroll
        for (int o = 16; o >= 1; o >>= 1) sum += __shfl_xor(sum, o, 32);
        const float inv = 1.0f / sum;
        _Float16* ab = attn16 + ((size_t)(bb * NHh + row) * Sn + q) * Sn;
#pragma unroll
        for (int i = 0; i < 8; ++i) ab[lane + 32 * i] = (_Float16)(e[i] * inv);
    }
}

// ---------------- ctx = attn @ V per (b,h)  (M=256, N=64, K=256) ----------------
__global__ __launch_bounds__(32) void k_ctx(const _Float16* __restrict__ attn16,
                                            const _Float16* __restrict__ vvT16,
                                            _Float16* __restrict__ ao16) {
    const int lane = threadIdx.x;
    const int qt = blockIdx.x, dt = blockIdx.y, bh = blockIdx.z;
    const _Float16* ab = attn16 + (size_t)bh * Sn * Sn;
    const _Float16* vb = vvT16 + (size_t)bh * Dh * Sn;
    v8f acc = {};
#pragma unroll
    for (int kk = 0; kk < Sn; kk += 32) {
        v16h a = load_a(ab, Sn, qt * 16, kk, lane);
        v16h b = load_b(vb, Sn, dt * 16, kk, lane);
        acc = wmma_f16(a, b, acc);
    }
    const int bb = bh / NHh, h = bh - bb * NHh;
    const int n = dt * 16 + (lane & 15);
    const int mb = (lane >> 4) * 8;
#pragma unroll
    for (int j = 0; j < 8; ++j) {
        const int m = qt * 16 + mb + j;
        ao16[((size_t)bb * Sn + m) * Hn + h * Dh + n] = (_Float16)acc[j];
    }
}

// ---------------- FF + residual (M=512, N=768, K=768) ----------------
__global__ __launch_bounds__(32) void k_ff(const _Float16* __restrict__ ao16,
                                           const _Float16* __restrict__ wffT,
                                           const float* __restrict__ bff,
                                           const float* __restrict__ inp,
                                           float* __restrict__ yres) {
    const int lane = threadIdx.x;
    const int mt = blockIdx.x, nt = blockIdx.y;
    v8f acc = {};
    for (int kk = 0; kk < Hn; kk += 32) {
        v16h a = load_a(ao16, Hn, mt * 16, kk, lane);
        v16h b = load_b(wffT, Hn, nt * 16, kk, lane);
        acc = wmma_f16(a, b, acc);
    }
    const int n = nt * 16 + (lane & 15);
    const int mb = (lane >> 4) * 8;
#pragma unroll
    for (int j = 0; j < 8; ++j) {
        const int m = mt * 16 + mb + j;
        const size_t idx = (size_t)m * Hn + n;
        yres[idx] = acc[j] + bff[n] + inp[idx];
    }
}

// ---------------- LayerNorm per row ----------------
__global__ __launch_bounds__(256) void k_ln(const float* __restrict__ yres,
                                            const float* __restrict__ g,
                                            const float* __restrict__ bta,
                                            float* __restrict__ out) {
    __shared__ float red[256];
    const int row = blockIdx.x, tid = threadIdx.x;
    const float* yr = yres + (size_t)row * Hn;
    float s = 0.f;
    for (int i = tid; i < Hn; i += 256) s += yr[i];
    red[tid] = s; __syncthreads();
    for (int o = 128; o > 0; o >>= 1) { if (tid < o) red[tid] += red[tid + o]; __syncthreads(); }
    const float mu = red[0] / Hn; __syncthreads();
    float v2 = 0.f;
    for (int i = tid; i < Hn; i += 256) { float d = yr[i] - mu; v2 += d * d; }
    red[tid] = v2; __syncthreads();
    for (int o = 128; o > 0; o >>= 1) { if (tid < o) red[tid] += red[tid + o]; __syncthreads(); }
    const float var = red[0] / Hn;
    const float rs = rsqrtf(var + 1e-12f);
    for (int i = tid; i < Hn; i += 256) {
        // final output is never re-read on-device: nontemporal store
        __builtin_nontemporal_store((yr[i] - mu) * rs * g[i] + bta[i], &out[(size_t)row * Hn + i]);
    }
}

static inline size_t align_up(size_t x) { return (x + 255) & ~(size_t)255; }

extern "C" void kernel_launch(void* const* d_in, const int* in_sizes, int n_in,
                              void* d_out, int out_size, void* d_ws, size_t ws_size,
                              hipStream_t stream) {
    const float* inp  = (const float*)d_in[0];
    const float* pos  = (const float*)d_in[1];
    const unsigned char* mask = (const unsigned char*)d_in[2];  // jnp bool -> 1 byte
    const float* Wq = (const float*)d_in[3];
    const float* bq = (const float*)d_in[4];
    const float* Wk = (const float*)d_in[5];
    const float* bk = (const float*)d_in[6];
    const float* Wv = (const float*)d_in[7];
    const float* bv = (const float*)d_in[8];
    const float* Wr = (const float*)d_in[9];
    // d_in[10] = br: adds a k-independent constant to each softmax row -> cancels exactly; unused.
    const float* u  = (const float*)d_in[11];
    const float* v  = (const float*)d_in[12];
    const float* Wff = (const float*)d_in[13];
    const float* bff = (const float*)d_in[14];
    const float* ln_g = (const float*)d_in[15];
    const float* ln_b = (const float*)d_in[16];
    float* out = (float*)d_out;

    // ---- carve workspace ----
    char* p = (char*)d_ws;
    auto carve = [&](size_t bytes) { char* r = p; p += align_up(bytes); return r; };
    _Float16* x16   = (_Float16*)carve((size_t)BS * Hn * 2);
    _Float16* wqT   = (_Float16*)carve((size_t)Hn * Hn * 2);
    _Float16* wkT   = (_Float16*)carve((size_t)Hn * Hn * 2);
    _Float16* wvT   = (_Float16*)carve((size_t)Hn * Hn * 2);
    _Float16* wffT  = (_Float16*)carve((size_t)Hn * Hn * 2);
    _Float16* wr16  = (_Float16*)carve((size_t)Hn * Hn * 2);
    _Float16* qu16  = (_Float16*)carve((size_t)Bn * NHh * Sn * Dh * 2);
    _Float16* qv16  = (_Float16*)carve((size_t)Bn * NHh * Sn * Dh * 2);
    _Float16* kk16  = (_Float16*)carve((size_t)Bn * NHh * Sn * Dh * 2);
    _Float16* vvT16 = (_Float16*)carve((size_t)Bn * NHh * Dh * Sn * 2);
    _Float16* w16   = (_Float16*)carve((size_t)Bn * NHh * Sn * Hn * 2);
    float*    acS   = (float*)   carve((size_t)Bn * NHh * Sn * Sn * 4);
    _Float16* at16  = (_Float16*)carve((size_t)Bn * NHh * Sn * Sn * 2);
    _Float16* ao16  = (_Float16*)carve((size_t)BS * Hn * 2);
    float*    yres  = (float*)   carve((size_t)BS * Hn * 4);
    (void)ws_size; (void)n_in; (void)in_sizes; (void)out_size;

    // ---- precision conversion ----
    k_conv<<<(BS * Hn + 255) / 256, 256, 0, stream>>>(inp, x16, BS * Hn);
    k_transconv<<<(Hn * Hn + 255) / 256, 256, 0, stream>>>(Wq, wqT);
    k_transconv<<<(Hn * Hn + 255) / 256, 256, 0, stream>>>(Wk, wkT);
    k_transconv<<<(Hn * Hn + 255) / 256, 256, 0, stream>>>(Wv, wvT);
    k_transconv<<<(Hn * Hn + 255) / 256, 256, 0, stream>>>(Wff, wffT);
    k_conv<<<(Hn * Hn + 255) / 256, 256, 0, stream>>>(Wr, wr16, Hn * Hn);

    // ---- pipeline ----
    k_qkv<<<dim3(BS / 16, Hn / 16, 3), 32, 0, stream>>>(x16, wqT, wkT, wvT, bq, bk, bv, u, v,
                                                        qu16, qv16, kk16, vvT16);
    k_wproj<<<dim3(BS / 16, Hn / 16, NHh), 32, 0, stream>>>(qv16, wr16, w16);
    k_ac<<<dim3(Sn / 16, Sn / 16, Bn * NHh), 32, 0, stream>>>(qu16, kk16, acS);
    k_bd_softmax<<<dim3(Sn, Bn), 256, 0, stream>>>(w16, pos, acS, mask, at16);
    k_ctx<<<dim3(Sn / 16, Dh / 16, Bn * NHh), 32, 0, stream>>>(at16, vvT16, ao16);
    k_ff<<<dim3(BS / 16, Hn / 16), 32, 0, stream>>>(ao16, wffT, bff, inp, yres);
    k_ln<<<BS, 256, 0, stream>>>(yres, ln_g, ln_b, out);
}
